// PairwiseAttentionBlock_4561255268372
// MI455X (gfx1250) — matrix operand order repaired and checked by reading, not compiled, not run on hardware
//
#include <hip/hip_runtime.h>
#include <hip/hip_bf16.h>
#include <stdint.h>

// ---------------------------------------------------------------------------
// PairwiseAttentionBlock for MI455X (gfx1250), wave32 + WMMA bf16.
// B=4, N=512, D=256, H=8, HD=32, P=3, FF=1024.
// ---------------------------------------------------------------------------

typedef __bf16 bf16_t;
typedef __attribute__((ext_vector_type(16))) __bf16 v16bf;
typedef __attribute__((ext_vector_type(8)))  float  v8f;

#define B_   4
#define N_   512
#define D_   256
#define H_   8
#define HD_  32
#define FF_  1024
#define TOK_ (B_ * N_)                 // 2048 tokens
#define QK_SCALE 0.17677669529663687f  // 1/sqrt(HD)

__device__ __forceinline__ int lane_id() { return (int)(threadIdx.x & 31); }

__device__ __forceinline__ v8f wmma_bf16(v16bf a, v16bf b, v8f c) {
  return __builtin_amdgcn_wmma_f32_16x16x32_bf16(false, a, false, b, (short)0, c,
                                                 false, false);
}

// A-tile (16x32 bf16, M x K) from row-major rows (stride in elements).
// ISA layout: lane m=l&15, half=l>>4; vgpr v<4 -> k=half*8+2v(+1); v>=4 -> 16+half*8+2(v-4)(+1)
// => dwords {half*4+0..3, 8+half*4+0..3} of the 16-dword row.
__device__ __forceinline__ v16bf load_a_tile(const bf16_t* row0, int stride, int kbase) {
  int l = lane_id(); int m = l & 15; int half = l >> 4;
  const uint32_t* r = (const uint32_t*)(row0 + (long)m * stride + kbase);
  union { v16bf v; uint4 q[2]; } A;
  A.q[0] = *(const uint4*)(r + half * 4);
  A.q[1] = *(const uint4*)(r + 8 + half * 4);
  return A.v;
}

// B-tile (32x16 bf16, K x N) from "Wt" storage: row n holds k-contiguous data.
// ISA layout: lane n=l&15, half=l>>4; vgpr v<4 -> k=half*8+2v(+1); v>=4 -> 16+half*8+2(v-4)(+1)
// => dwords {half*4+0..3, 8+half*4+0..3} of the 16-dword row.
__device__ __forceinline__ v16bf load_b_tile(const bf16_t* colrow0, int stride, int kbase) {
  int l = lane_id(); int n = l & 15; int half = l >> 4;
  const uint32_t* r = (const uint32_t*)(colrow0 + (long)n * stride + kbase);
  union { v16bf v; uint4 q[2]; } Bv;
  Bv.q[0] = *(const uint4*)(r + half * 4);
  Bv.q[1] = *(const uint4*)(r + 8 + half * 4);
  return Bv.v;
}

// ---------------------------------------------------------------------------
// Pack: bf16 copies, transposed weights (Wt[n][k]), zero-padded Wp2T[16][256].
// ---------------------------------------------------------------------------
__global__ void pack_kernel(const float* x, const float* Wqkv, const float* Wout,
                            const float* Wf1, const float* Wf2, const float* Wp2,
                            bf16_t* xb, bf16_t* WqkvT, bf16_t* WoutT,
                            bf16_t* Wf1T, bf16_t* Wf2T, bf16_t* Wp2T) {
  int stride = gridDim.x * blockDim.x;
  int tid = blockIdx.x * blockDim.x + threadIdx.x;
  for (int i = tid; i < TOK_ * D_; i += stride) xb[i] = (bf16_t)x[i];
  for (int i = tid; i < 3 * D_ * D_; i += stride) {
    int nn = i / D_, kk = i % D_;
    WqkvT[i] = (bf16_t)Wqkv[kk * (3 * D_) + nn];
  }
  for (int i = tid; i < D_ * D_; i += stride) {
    int nn = i / D_, kk = i % D_;
    WoutT[i] = (bf16_t)Wout[kk * D_ + nn];
  }
  for (int i = tid; i < FF_ * D_; i += stride) {
    int nn = i / D_, kk = i % D_;
    Wf1T[i] = (bf16_t)Wf1[kk * FF_ + nn];
  }
  for (int i = tid; i < D_ * FF_; i += stride) {
    int nn = i / FF_, kk = i % FF_;
    Wf2T[i] = (bf16_t)Wf2[kk * D_ + nn];
  }
  for (int i = tid; i < 16 * D_; i += stride) {
    int nn = i / D_, kk = i % D_;
    Wp2T[i] = (nn < H_) ? (bf16_t)Wp2[kk * H_ + nn] : (bf16_t)0.0f;
  }
}

// ---------------------------------------------------------------------------
// Pairwise bias MLP, fused: h = relu(pair @ Wp1 + bp1) built directly in the
// WMMA A-register layout (16 pairs x 256 channels), then h @ Wp2 on the
// matrix core. Writes bias[B,H,N,N] f32. One wave = 16 (i,j) pairs.
// ---------------------------------------------------------------------------
__global__ void bias_kernel(const float* __restrict__ pair, const float* __restrict__ Wp1,
                            const float* __restrict__ bp1, const bf16_t* __restrict__ Wp2T,
                            const float* __restrict__ bp2, float* __restrict__ biasbuf) {
  __shared__ float w1s[3 * D_];
  __shared__ float b1s[D_];
  for (int i = threadIdx.x; i < 3 * D_; i += blockDim.x) w1s[i] = Wp1[i];
  for (int i = threadIdx.x; i < D_; i += blockDim.x) b1s[i] = bp1[i];
  __syncthreads();

  int wave = threadIdx.x >> 5;
  int l = lane_id(); int m = l & 15; int half = l >> 4; int n = m;
  long tile = (long)blockIdx.x * 4 + wave;
  long pbase = tile * 16;
  const float* pf = pair + (pbase + m) * 3;
  float p0 = pf[0], p1 = pf[1], p2 = pf[2];

  v8f acc = {};
  for (int ks = 0; ks < 8; ++ks) {
    union { v16bf v; bf16_t e[16]; } A;
#pragma unroll
    for (int e = 0; e < 16; ++e) {
      int vg = e >> 1, odd = e & 1;
      int k = (vg < 4) ? (half * 8 + 2 * vg + odd) : (16 + half * 8 + 2 * (vg - 4) + odd);
      int c = ks * 32 + k;
      float hval = fmaf(p0, w1s[c], fmaf(p1, w1s[D_ + c], fmaf(p2, w1s[2 * D_ + c], b1s[c])));
      A.e[e] = (bf16_t)fmaxf(hval, 0.0f);
    }
    v16bf bw = load_b_tile(Wp2T, D_, ks * 32);
    acc = wmma_bf16(A.v, bw, acc);
  }

  if (n < H_) {
    float bb2 = bp2[n];
#pragma unroll
    for (int r = 0; r < 8; ++r) {
      long pp = pbase + r + 8 * half;
      long bidx = pp / ((long)N_ * N_);
      long ij = pp % ((long)N_ * N_);
      int ii = (int)(ij / N_), jj = (int)(ij % N_);
      biasbuf[(((long)(bidx * H_ + n) * N_ + ii) * N_ + jj)] = acc[r] + bb2;
    }
  }
}

// ---------------------------------------------------------------------------
// Generic bf16 WMMA GEMM: C[M,N] = A[M,K] @ Bt[n][k].
// Each wave owns a 16x64 strip: the A fragment is loaded once per k-step and
// reused across 4 B tiles (4 accumulators) -> 1 A-load : 4 (B-load + wmma).
// EPI 0: QKV scatter (q scaled, k, v-transposed) | 1: +bias -> f32 | 2: +bias, gelu -> bf16
// ---------------------------------------------------------------------------
template <int EPI>
__global__ void gemm_kernel(const bf16_t* __restrict__ A, const bf16_t* __restrict__ Bt,
                            int M, int N, int K, const float* __restrict__ bias,
                            float* __restrict__ Cf, bf16_t* __restrict__ Cb,
                            bf16_t* __restrict__ qh, bf16_t* __restrict__ kh,
                            bf16_t* __restrict__ vt) {
  int wave = threadIdx.x >> 5;
  int row0 = blockIdx.y * 16;
  int col0 = blockIdx.x * 256 + wave * 64;
  const bf16_t* arow = A + (long)row0 * K;
  const bf16_t* brow = Bt + (long)col0 * K;
  v8f acc[4] = {v8f{}, v8f{}, v8f{}, v8f{}};
  for (int kc = 0; kc < K; kc += 32) {
    __builtin_prefetch(arow + kc + 128, 0, 1);  // global_prefetch hint
    v16bf a = load_a_tile(arow, K, kc);
#pragma unroll
    for (int t = 0; t < 4; ++t) {
      v16bf b = load_b_tile(brow + (long)t * 16 * K, K, kc);
      acc[t] = wmma_bf16(a, b, acc[t]);
    }
  }
  int l = lane_id(); int n = l & 15; int half = l >> 4;
#pragma unroll
  for (int t = 0; t < 4; ++t) {
    int col = col0 + t * 16 + n;
#pragma unroll
    for (int r = 0; r < 8; ++r) {
      int row = row0 + r + 8 * half;
      float v = acc[t][r];
      if (EPI == 0) {
        int bidx = row >> 9, i = row & (N_ - 1);
        if (col < D_) {
          int h = col >> 5, d = col & 31;
          qh[(((long)(bidx * H_ + h) * N_ + i) << 5) + d] = (bf16_t)(v * QK_SCALE);
        } else if (col < 2 * D_) {
          int c = col - D_; int h = c >> 5, d = c & 31;
          kh[(((long)(bidx * H_ + h) * N_ + i) << 5) + d] = (bf16_t)v;
        } else {
          int c = col - 2 * D_; int h = c >> 5, d = c & 31;
          vt[((long)(bidx * H_ + h) * HD_ + d) * N_ + i] = (bf16_t)v;
        }
      } else if (EPI == 1) {
        Cf[(long)row * N + col] = v + bias[col];
      } else if (EPI == 2) {
        float xv = v + bias[col];
        float g = 0.5f * xv * (1.0f + erff(xv * 0.70710678118654752f));
        Cb[(long)row * N + col] = (bf16_t)g;
      }
    }
  }
}

// ---------------------------------------------------------------------------
// Flash attention per (b, h, 16-row i-tile). HD=32 => QK^T is one WMMA per
// 16-key tile; P restaged through LDS (A-layout) for the PV WMMAs.
// ---------------------------------------------------------------------------
__global__ void attn_kernel(const bf16_t* __restrict__ qh, const bf16_t* __restrict__ kh,
                            const bf16_t* __restrict__ vt, const float* __restrict__ biasbuf,
                            const uint8_t* __restrict__ mask, bf16_t* __restrict__ attn_out) {
  __shared__ __align__(16) bf16_t Plds[4][16 * 32];
  int wave = threadIdx.x >> 5;
  int job = blockIdx.x * 4 + wave;           // B*H*(N/16) = 1024 jobs
  int b = job >> 8;
  int rem = job & 255;
  int h = rem >> 5;
  int i0 = (rem & 31) * 16;

  int l = lane_id(); int n = l & 15; int half = l >> 4;
  const bf16_t* qbase = qh + ((long)(b * H_ + h) * N_ + i0) * HD_;
  const bf16_t* kbase = kh + (long)(b * H_ + h) * N_ * HD_;
  const bf16_t* vbase = vt + (long)(b * H_ + h) * HD_ * N_;
  const float* brow0 = biasbuf + ((long)(b * H_ + h) * N_ + i0) * N_;
  const uint8_t* mrow = mask + b * N_;

  v16bf aQ = load_a_tile(qbase, HD_, 0);     // q already scaled by 1/sqrt(HD)
  v8f o0 = {}, o1 = {};
  float mrun[8], lrun[8];
#pragma unroll
  for (int r = 0; r < 8; ++r) { mrun[r] = -3.0e38f; lrun[r] = 0.0f; }
  bf16_t* pl = Plds[wave];

  for (int jc = 0; jc < N_; jc += 32) {
    v16bf bK0 = load_b_tile(kbase + (long)jc * HD_, HD_, 0);
    v16bf bK1 = load_b_tile(kbase + (long)(jc + 16) * HD_, HD_, 0);
    v8f s0 = {}, s1 = {};
    s0 = wmma_bf16(aQ, bK0, s0);
    s1 = wmma_bf16(aQ, bK1, s1);

    bool vm0 = mrow[jc + n] != 0;
    bool vm1 = mrow[jc + 16 + n] != 0;
    float t0[8], t1[8];
#pragma unroll
    for (int r = 0; r < 8; ++r) {
      int m = r + 8 * half;
      float a0 = s0[r] + brow0[(long)m * N_ + jc + n];
      float a1 = s1[r] + brow0[(long)m * N_ + jc + 16 + n];
      t0[r] = vm0 ? a0 : -1.0e9f;
      t1[r] = vm1 ? a1 : -1.0e9f;
    }
#pragma unroll
    for (int r = 0; r < 8; ++r) {
      float mx = fmaxf(t0[r], t1[r]);
      mx = fmaxf(mx, __shfl_xor(mx, 1, 16));
      mx = fmaxf(mx, __shfl_xor(mx, 2, 16));
      mx = fmaxf(mx, __shfl_xor(mx, 4, 16));
      mx = fmaxf(mx, __shfl_xor(mx, 8, 16));
      float mnew = fmaxf(mrun[r], mx);
      float p0 = __expf(t0[r] - mnew);
      float p1 = __expf(t1[r] - mnew);
      float rs = p0 + p1;
      rs += __shfl_xor(rs, 1, 16);
      rs += __shfl_xor(rs, 2, 16);
      rs += __shfl_xor(rs, 4, 16);
      rs += __shfl_xor(rs, 8, 16);
      float alpha = __expf(mrun[r] - mnew);
      lrun[r] = lrun[r] * alpha + rs;
      mrun[r] = mnew;
      o0[r] *= alpha;
      o1[r] *= alpha;
      int m = r + 8 * half;
      pl[m * 32 + n] = (bf16_t)p0;          // ds_store, in-order within wave
      pl[m * 32 + 16 + n] = (bf16_t)p1;
    }
    asm volatile("" ::: "memory");
    v16bf aP = load_a_tile(pl, 32, 0);      // ds_load, same-wave ordered
    v16bf bV0 = load_b_tile(vbase, N_, jc);
    v16bf bV1 = load_b_tile(vbase + 16 * N_, N_, jc);
    o0 = wmma_bf16(aP, bV0, o0);
    o1 = wmma_bf16(aP, bV1, o1);
  }

#pragma unroll
  for (int r = 0; r < 8; ++r) {
    int m = r + 8 * half;
    float inv = 1.0f / lrun[r];
    long row = (long)b * N_ + i0 + m;
    attn_out[row * D_ + h * HD_ + n] = (bf16_t)(o0[r] * inv);
    attn_out[row * D_ + h * HD_ + 16 + n] = (bf16_t)(o1[r] * inv);
  }
}

// ---------------------------------------------------------------------------
// Residual + LayerNorm, one wave per 256-wide row.
// ---------------------------------------------------------------------------
__global__ void ln_kernel(const float* __restrict__ xa, const float* __restrict__ xb,
                          const float* __restrict__ g, const float* __restrict__ bb,
                          float* __restrict__ outf, bf16_t* __restrict__ outb) {
  int wave = threadIdx.x >> 5;
  int row = blockIdx.x * 4 + wave;
  int l = lane_id();
  const float* pa = xa + (long)row * D_;
  const float* pb = xb + (long)row * D_;
  float v[8], s = 0.0f, sq = 0.0f;
#pragma unroll
  for (int e = 0; e < 8; ++e) {
    int c = l + e * 32;
    float t = pa[c] + pb[c];
    v[e] = t; s += t; sq += t * t;
  }
#pragma unroll
  for (int off = 1; off < 32; off <<= 1) {
    s += __shfl_xor(s, off, 32);
    sq += __shfl_xor(sq, off, 32);
  }
  float mean = s * (1.0f / D_);
  float var = sq * (1.0f / D_) - mean * mean;
  float inv = rsqrtf(var + 1e-5f);
#pragma unroll
  for (int e = 0; e < 8; ++e) {
    int c = l + e * 32;
    float y = (v[e] - mean) * inv * g[c] + bb[c];
    outf[(long)row * D_ + c] = y;
    if (outb) outb[(long)row * D_ + c] = (bf16_t)y;
  }
}

// ---------------------------------------------------------------------------
extern "C" void kernel_launch(void* const* d_in, const int* in_sizes, int n_in,
                              void* d_out, int out_size, void* d_ws, size_t ws_size,
                              hipStream_t stream) {
  (void)in_sizes; (void)n_in; (void)out_size; (void)ws_size;
  const float*   x    = (const float*)d_in[0];
  const uint8_t* mask = (const uint8_t*)d_in[1];   // jax bool -> 1 byte
  const float*   pair = (const float*)d_in[2];
  const float*   Wqkv = (const float*)d_in[3];
  const float*   Wout = (const float*)d_in[4];
  const float*   bout = (const float*)d_in[5];
  const float*   Wp1  = (const float*)d_in[6];
  const float*   bp1  = (const float*)d_in[7];
  const float*   Wp2  = (const float*)d_in[8];
  const float*   bp2  = (const float*)d_in[9];
  const float*   Wf1  = (const float*)d_in[10];
  const float*   bf1  = (const float*)d_in[11];
  const float*   Wf2  = (const float*)d_in[12];
  const float*   bf2  = (const float*)d_in[13];
  const float*   g1   = (const float*)d_in[14];
  const float*   b1   = (const float*)d_in[15];
  const float*   g2   = (const float*)d_in[16];
  const float*   b2   = (const float*)d_in[17];

  char* ws = (char*)d_ws;
  size_t off = 0;
  auto carve = [&](size_t bytes) -> char* {
    char* p = ws + off;
    off = (off + bytes + 255) & ~(size_t)255;
    return p;
  };
  bf16_t* xb      = (bf16_t*)carve((size_t)TOK_ * D_ * 2);
  bf16_t* WqkvT   = (bf16_t*)carve((size_t)3 * D_ * D_ * 2);
  bf16_t* WoutT   = (bf16_t*)carve((size_t)D_ * D_ * 2);
  bf16_t* Wf1T    = (bf16_t*)carve((size_t)FF_ * D_ * 2);
  bf16_t* Wf2T    = (bf16_t*)carve((size_t)D_ * FF_ * 2);
  bf16_t* Wp2T    = (bf16_t*)carve((size_t)16 * D_ * 2);
  bf16_t* qh      = (bf16_t*)carve((size_t)B_ * H_ * N_ * HD_ * 2);
  bf16_t* kh      = (bf16_t*)carve((size_t)B_ * H_ * N_ * HD_ * 2);
  bf16_t* vtb     = (bf16_t*)carve((size_t)B_ * H_ * HD_ * N_ * 2);
  float*  biasbuf = (float*)carve((size_t)B_ * H_ * N_ * N_ * 4);
  bf16_t* attn_o  = (bf16_t*)carve((size_t)TOK_ * D_ * 2);
  float*  tmp1    = (float*)carve((size_t)TOK_ * D_ * 4);
  float*  x1      = (float*)carve((size_t)TOK_ * D_ * 4);
  bf16_t* x1b     = (bf16_t*)carve((size_t)TOK_ * D_ * 2);
  bf16_t* hff     = (bf16_t*)carve((size_t)TOK_ * FF_ * 2);
  float*  tmp2    = (float*)carve((size_t)TOK_ * D_ * 4);

  // 1) bf16 packing + weight transposes
  pack_kernel<<<512, 256, 0, stream>>>(x, Wqkv, Wout, Wf1, Wf2, Wp2,
                                       xb, WqkvT, WoutT, Wf1T, Wf2T, Wp2T);
  // 2) pairwise bias MLP on the matrix core -> bias[B,H,N,N]
  bias_kernel<<<(B_ * N_ * N_ / 16) / 4, 128, 0, stream>>>(pair, Wp1, bp1, Wp2T, bp2, biasbuf);
  // 3) QKV projection, scatter to per-head q/k and transposed v  (N=768 -> grid.x=3)
  gemm_kernel<0><<<dim3(3, TOK_ / 16), 128, 0, stream>>>(
      xb, WqkvT, TOK_, 3 * D_, D_, nullptr, nullptr, nullptr, qh, kh, vtb);
  // 4) flash attention with pair bias + key mask
  attn_kernel<<<(B_ * H_ * (N_ / 16)) / 4, 128, 0, stream>>>(qh, kh, vtb, biasbuf, mask, attn_o);
  // 5) output projection (+bout)  (N=256 -> grid.x=1)
  gemm_kernel<1><<<dim3(1, TOK_ / 16), 128, 0, stream>>>(
      attn_o, WoutT, TOK_, D_, D_, bout, tmp1, nullptr, nullptr, nullptr, nullptr);
  // 6) residual + LN1 -> x1 (f32 + bf16)
  ln_kernel<<<TOK_ / 4, 128, 0, stream>>>(x, tmp1, g1, b1, x1, x1b);
  // 7) FFN up (+bf1, exact GELU) -> bf16  (N=1024 -> grid.x=4)
  gemm_kernel<2><<<dim3(4, TOK_ / 16), 128, 0, stream>>>(
      x1b, Wf1T, TOK_, FF_, D_, bf1, nullptr, hff, nullptr, nullptr, nullptr);
  // 8) FFN down (+bf2)  (N=256 -> grid.x=1)
  gemm_kernel<1><<<dim3(1, TOK_ / 16), 128, 0, stream>>>(
      hff, Wf2T, TOK_, D_, FF_, bf2, tmp2, nullptr, nullptr, nullptr, nullptr);
  // 9) residual + LN2 -> d_out
  ln_kernel<<<TOK_ / 4, 128, 0, stream>>>(x1, tmp2, g2, b2, (float*)d_out, nullptr);
}
